// TangentSpaceHead_5746666242178
// MI455X (gfx1250) — matrix operand order repaired and checked by reading, not compiled, not run on hardware
//
#include <hip/hip_runtime.h>
#include <hip/hip_bf16.h>
#include <math.h>

typedef __attribute__((ext_vector_type(16))) _Float16 v16h;
typedef __attribute__((ext_vector_type(8)))  float    v8f;
typedef __attribute__((ext_vector_type(2)))  float    v2f;

#define NF   64
#define NC   128
#define NT   2048
#define TT   64
#define XS_STR 72
#define XF_STR 72
#define AST  65
#define OUTD 2080

// map flat upper-triangular index u -> (r,c), r<=c, row-major triu order
__device__ __forceinline__ void tri_rc(int u, int& r, int& c) {
  int rr = (int)((129.0f - sqrtf(16641.0f - 8.0f * (float)u)) * 0.5f);
  if (rr < 0) rr = 0;
  if (rr > 63) rr = 63;
  while (rr > 0 && u < rr * (129 - rr) / 2) --rr;
  while (rr < 63 && u >= (rr + 1) * (128 - rr) / 2) ++rr;
  r = rr;
  c = rr + (u - rr * (129 - rr) / 2);
}

// round-robin (circle method) disjoint pair schedule: 63 rounds x 32 pairs
__device__ __forceinline__ void jpair(int rnd, int k, int& p, int& q) {
  if (k == 0) { p = 63; q = rnd; }
  else { p = (rnd + k) % 63; q = (rnd - k + 63) % 63; }
}

__global__ __launch_bounds__(256) void tshead_kernel(
    const float* __restrict__ x, const float* __restrict__ w_proj,
    const float* __restrict__ w_cls, float* __restrict__ out, int B) {

  __shared__ float bufA[NF * AST];          // Gram -> cov -> Jacobi A
  __shared__ union {
    struct {                                 // projection/cov phase
      _Float16 wp[NF * NC];                  // w_proj f16 [f][c]
      _Float16 xs[NC * XS_STR];              // x tile f16 [c][t]
      _Float16 xf[NF * XF_STR];              // xf tile f16 [f][t]
    } p;
    struct {                                 // eigen phase
      float V[NF * AST];                     // eigenvectors
      float lc[NF * AST];                    // log-cov
      float ld[NF];                          // log eigenvalues
    } e;
  } su;
  __shared__ float smean[NF];
  __shared__ float crot[32], srot[32];
  __shared__ float red[4 * 256];

  const int b    = blockIdx.x;
  const int tid  = threadIdx.x;
  const int wave = tid >> 5;
  const int lane = tid & 31;
  const int half = lane >> 4;
  const int l16  = lane & 15;

  // ---- stage w_proj (64x128 f32 -> f16 LDS) ----
  {
    const float4* w4 = (const float4*)w_proj;
#pragma unroll
    for (int j = 0; j < 8; ++j) {
      int i = tid + 256 * j;                 // 0..2047 float4s
      float4 v = w4[i];
      _Float16* d = &su.p.wp[i * 4];
      d[0] = (_Float16)v.x; d[1] = (_Float16)v.y;
      d[2] = (_Float16)v.z; d[3] = (_Float16)v.w;
    }
  }
  if (tid < NF) smean[tid] = 0.f;
  v8f g0 = {}; v8f g1 = {};                  // Gram accumulators (2 tiles/wave)

  const float* xb = x + (size_t)b * NC * NT;
  const int crow = tid >> 4;                 // 0..15 : row of x tile this thread loads
  const int ct4  = (tid & 15) << 2;          // 0,4,..60 : col offset (float4)
  const float* xrow = xb + (size_t)crow * NT + ct4;

  // software-pipeline: stage[] holds the NEXT tile's global data in registers
  float4 stage[8];
#pragma unroll
  for (int j = 0; j < 8; ++j)                // prologue: tile t0 = 0
    stage[j] = *(const float4*)(xrow + 512 * j);   // 4 rows per j (64 rows apart? no: see below)

  __syncthreads();

  // NOTE on stage mapping: thread loads 8 float4s, one per "row group":
  // element j covers rows crow + 128*j? 128 rows total handled by 256 threads
  // at 16 float4/row -> each thread owns rows {crow} only if 256/16 = 16 thr/row.
  // 256 threads * 8 float4 = 2048 float4 = full 128x64 tile:
  // j-th piece is row (crow) of slab j? Rows are 128, threads cover 16 rows per j? No:
  // threads 0..255 at 16/row cover rows 0..15 for j=0 -> use stride 16 rows per j.
  // stage[j] = x[row = crow + ... ] : implemented via xrow + j * (16 rows * NT)
  // (the initializer above uses 512*j = 16*NT*... NT=2048 so 16 rows = 32768 floats;
  //  512*j is wrong -- fixed below by reloading with the correct stride.)

  // ==================== projection + Gram over T tiles ====================
  for (int t0 = 0; t0 < NT; t0 += TT) {
    if (t0 == 0) {
      // redo prologue loads with correct addressing (row block stride = 16 rows)
#pragma unroll
      for (int j = 0; j < 8; ++j)
        stage[j] = *(const float4*)(xb + (size_t)(crow & 15) * NT +
                                    (size_t)((crow >> 4) + j * 8) * 0);
      // simple, correct mapping: thread handles rows r = tid>>4 (0..15) + 16*j
#pragma unroll
      for (int j = 0; j < 8; ++j) {
        int c = (tid >> 4) + 16 * j;         // 0..127
        stage[j] = *(const float4*)(xb + (size_t)c * NT + t0 + ct4);
      }
    }
    // store staged tile -> xs (f16), waits on outstanding loads automatically
#pragma unroll
    for (int j = 0; j < 8; ++j) {
      int c = (tid >> 4) + 16 * j;
      _Float16* d = &su.p.xs[c * XS_STR + ct4];
      d[0] = (_Float16)stage[j].x; d[1] = (_Float16)stage[j].y;
      d[2] = (_Float16)stage[j].z; d[3] = (_Float16)stage[j].w;
    }
    __syncthreads();

    // issue NEXT tile's global loads now; they overlap the WMMA work below
    if (t0 + TT < NT) {
#pragma unroll
      for (int j = 0; j < 8; ++j) {
        int c = (tid >> 4) + 16 * j;
        stage[j] = *(const float4*)(xb + (size_t)c * NT + (t0 + TT) + ct4);
        if (t0 + 2 * TT < NT)
          __builtin_prefetch(xb + (size_t)c * NT + (t0 + 2 * TT) + ct4, 0, 1);
      }
    }

    // projection: xf(64x64) = wp(64x128) @ xs(128x64), 16 tiles, 2 per wave
    for (int tt = 0; tt < 2; ++tt) {
      int tile = wave * 2 + tt;
      int mi = tile >> 2, ni = tile & 3;
      v8f acc = {};
      for (int ks = 0; ks < 4; ++ks) {
        v16h a, bm;
#pragma unroll
        for (int e = 0; e < 16; ++e) {       // A 16x32 f16 layout
          int k = 32 * ks + (e & 7) + ((e >> 3) << 4) + (half << 3);
          a[e] = su.p.wp[(16 * mi + l16) * NC + k];
        }
#pragma unroll
        for (int e = 0; e < 16; ++e) {       // B 32x16 f16 layout
          int k = 32 * ks + (e & 7) + ((e >> 3) << 4) + (half << 3);
          bm[e] = su.p.xs[k * XS_STR + 16 * ni + l16];
        }
        acc = __builtin_amdgcn_wmma_f32_16x16x32_f16(
            false, a, false, bm, (short)0, acc, false, false);
      }
#pragma unroll
      for (int rI = 0; rI < 8; ++rI)         // C/D layout: M=r+8*half, N=l16
        su.p.xf[(16 * mi + rI + 8 * half) * XF_STR + 16 * ni + l16] =
            (_Float16)acc[rI];
    }
    __syncthreads();

    // per-row sums for mean (single writer per row)
    if (tid < NF) {
      float s = 0.f;
#pragma unroll 8
      for (int j = 0; j < TT; ++j) s += (float)su.p.xf[tid * XF_STR + j];
      smean[tid] += s;
    }

    // Gram += xf @ xf^T  (A rows = f, B cols = g, K = t)
    for (int tt = 0; tt < 2; ++tt) {
      int tile = wave * 2 + tt;
      int gi = tile >> 2, gj = tile & 3;
      v8f g = tt ? g1 : g0;
      for (int ks = 0; ks < 2; ++ks) {
        v16h a, bm;
#pragma unroll
        for (int e = 0; e < 16; ++e) {
          int k = 32 * ks + (e & 7) + ((e >> 3) << 4) + (half << 3);
          a[e] = su.p.xf[(16 * gi + l16) * XF_STR + k];
        }
#pragma unroll
        for (int e = 0; e < 16; ++e) {       // B[k][n] = xf[n][k]
          int k = 32 * ks + (e & 7) + ((e >> 3) << 4) + (half << 3);
          bm[e] = su.p.xf[(16 * gj + l16) * XF_STR + k];
        }
        g = __builtin_amdgcn_wmma_f32_16x16x32_f16(
            false, a, false, bm, (short)0, g, false, false);
      }
      if (tt) g1 = g; else g0 = g;
    }
    // no barrier here: next iteration's xs store targets a different buffer
    // than gram/mean reads (xf); the barrier after that store orders xs reads.
  }
  __syncthreads();

  // ---- dump Gram tiles to LDS ----
  for (int tt = 0; tt < 2; ++tt) {
    int tile = wave * 2 + tt;
    int gi = tile >> 2, gj = tile & 3;
    v8f g = tt ? g1 : g0;
#pragma unroll
    for (int rI = 0; rI < 8; ++rI)
      bufA[(16 * gi + rI + 8 * half) * AST + 16 * gj + l16] = g[rI];
  }
  __syncthreads();

  // ---- finalize cov: symmetrize, mean-center, /(T-1), +eps*I ----
  for (int u = tid; u < OUTD; u += 256) {
    int r, c; tri_rc(u, r, c);
    float raw = 0.5f * (bufA[r * AST + c] + bufA[c * AST + r]);
    float v = (raw - smean[r] * smean[c] * (1.0f / (float)NT)) *
              (1.0f / (float)(NT - 1));
    if (r == c) v += 1e-4f;
    bufA[r * AST + c] = v;
    bufA[c * AST + r] = v;
  }
  for (int i = tid; i < NF * AST; i += 256) su.e.V[i] = 0.f;
  __syncthreads();
  if (tid < NF) su.e.V[tid * AST + tid] = 1.f;
  __syncthreads();

  // ==================== cyclic Jacobi eigensolver ====================
  for (int sweep = 0; sweep < 8; ++sweep) {
    for (int rnd = 0; rnd < 63; ++rnd) {
      if (tid < 32) {                        // rotation params, disjoint pairs
        int p, q; jpair(rnd, tid, p, q);
        float app = bufA[p * AST + p];
        float aqq = bufA[q * AST + q];
        float apq = bufA[p * AST + q];
        float cth = 1.f, sth = 0.f;
        if (fabsf(apq) > 1e-12f * (fabsf(app) + fabsf(aqq) + 1e-30f)) {
          float tau = (aqq - app) / (2.f * apq);
          float t = copysignf(1.f, tau) / (fabsf(tau) + sqrtf(1.f + tau * tau));
          cth = rsqrtf(1.f + t * t);
          sth = t * cth;
        }
        crot[tid] = cth; srot[tid] = sth;
      }
      __syncthreads();
      // column updates: A <- A*J, V <- V*J   (32 pairs x 64 rows)
      for (int j = 0; j < 8; ++j) {
        int idx = tid + 256 * j;
        int pr = idx >> 6, i = idx & 63;
        int p, q; jpair(rnd, pr, p, q);
        float cth = crot[pr], sth = srot[pr];
        float aip = bufA[i * AST + p], aiq = bufA[i * AST + q];
        bufA[i * AST + p] = cth * aip - sth * aiq;
        bufA[i * AST + q] = sth * aip + cth * aiq;
        float vip = su.e.V[i * AST + p], viq = su.e.V[i * AST + q];
        su.e.V[i * AST + p] = cth * vip - sth * viq;
        su.e.V[i * AST + q] = sth * vip + cth * viq;
      }
      __syncthreads();
      // row updates: A <- J^T*A
      for (int j = 0; j < 8; ++j) {
        int idx = tid + 256 * j;
        int pr = idx >> 6, col = idx & 63;
        int p, q; jpair(rnd, pr, p, q);
        float cth = crot[pr], sth = srot[pr];
        float apj = bufA[p * AST + col], aqj = bufA[q * AST + col];
        bufA[p * AST + col] = cth * apj - sth * aqj;
        bufA[q * AST + col] = sth * apj + cth * aqj;
      }
      __syncthreads();
    }
  }

  // ---- log of clipped eigenvalues ----
  if (tid < NF) su.e.ld[tid] = logf(fmaxf(bufA[tid * AST + tid], 1e-4f));
  __syncthreads();

  // ==================== log_cov = V diag(ld) V^T via fp32 WMMA ====================
  for (int tt = 0; tt < 2; ++tt) {
    int tile = wave * 2 + tt;
    int mi = tile >> 2, ni = tile & 3;
    v8f acc = {};
    for (int ks = 0; ks < 16; ++ks) {
      v2f a2, b2;
#pragma unroll
      for (int e = 0; e < 2; ++e) {          // A 16x4 f32: K = e + 2*half
        int k = 4 * ks + e + 2 * half;
        a2[e] = su.e.V[(16 * mi + l16) * AST + k] * su.e.ld[k];
        b2[e] = su.e.V[(16 * ni + l16) * AST + k];   // B[k][n] = V[n][k]
      }
      acc = __builtin_amdgcn_wmma_f32_16x16x4_f32(
          false, a2, false, b2, (short)0, acc, false, false);
    }
#pragma unroll
    for (int rI = 0; rI < 8; ++rI)
      su.e.lc[(16 * mi + rI + 8 * half) * AST + 16 * ni + l16] = acc[rI];
  }
  __syncthreads();

  // ==================== vectorize + classifier ====================
  float* out_logits = out;
  float* out_ts = out + 2 * (size_t)B;
  const float* w0 = w_cls;
  const float* w1 = w_cls + OUTD;
  float sq0 = 0.f, sq1 = 0.f, d0 = 0.f, d1 = 0.f;
  for (int u = tid; u < OUTD; u += 256) {
    int r, c; tri_rc(u, r, c);
    float v = 0.5f * (su.e.lc[r * AST + c] + su.e.lc[c * AST + r]);
    if (r != c) v *= 1.41421356237309515f;
    out_ts[(size_t)b * OUTD + u] = v;
    float a0 = w0[u], a1 = w1[u];
    sq0 += a0 * a0; sq1 += a1 * a1;
    d0 += v * a0;  d1 += v * a1;
  }
  red[tid] = sq0; red[256 + tid] = sq1; red[512 + tid] = d0; red[768 + tid] = d1;
  __syncthreads();
  for (int off = 128; off > 0; off >>= 1) {
    if (tid < off) {
      red[tid]       += red[tid + off];
      red[256 + tid] += red[256 + tid + off];
      red[512 + tid] += red[512 + tid + off];
      red[768 + tid] += red[768 + tid + off];
    }
    __syncthreads();
  }
  if (tid == 0) {
    float n0 = sqrtf(red[0]),   n1 = sqrtf(red[256]);
    float s0 = (n0 > 0.25f) ? 0.25f / (n0 + 1e-7f) : 1.f;
    float s1 = (n1 > 0.25f) ? 0.25f / (n1 + 1e-7f) : 1.f;
    out_logits[b * 2 + 0] = red[512] * s0;
    out_logits[b * 2 + 1] = red[768] * s1;
  }
}

extern "C" void kernel_launch(void* const* d_in, const int* in_sizes, int n_in,
                              void* d_out, int out_size, void* d_ws, size_t ws_size,
                              hipStream_t stream) {
  const float* x      = (const float*)d_in[0];
  const float* w_proj = (const float*)d_in[1];
  const float* w_cls  = (const float*)d_in[2];
  float* out = (float*)d_out;
  int B = in_sizes[0] / (NC * NT);   // 512
  tshead_kernel<<<dim3(B), dim3(256), 0, stream>>>(x, w_proj, w_cls, out, B);
}
